// VariationalQuantizationLayer_80290118631474
// MI455X (gfx1250) — matrix operand order repaired and checked by reading, not compiled, not run on hardware
//
#include <hip/hip_runtime.h>
#include <math.h>

typedef __attribute__((ext_vector_type(16))) _Float16 v16h;
typedef __attribute__((ext_vector_type(8)))  _Float16 v8h;
typedef __attribute__((ext_vector_type(8)))  float    v8f;

__device__ __forceinline__ v16h combine16(v8h lo, v8h hi_) {
    v16h r;
    #pragma unroll
    for (int i = 0; i < 8; ++i) { r[i] = lo[i]; r[8 + i] = hi_[i]; }
    return r;
}

// ---------------------------------------------------------------------------
// emb0[k, c] = emb_table[k, c] + PE(k, c), 64 x 512 ; writes f32 + f16
// ---------------------------------------------------------------------------
__global__ void build_emb0_kernel(const float* __restrict__ tab,
                                  float* __restrict__ emb0,
                                  _Float16* __restrict__ emb0h) {
    int i = blockIdx.x * blockDim.x + threadIdx.x;
    if (i >= 64 * 512) return;
    int k = i >> 9;
    int c = i & 511;
    int iv = c & ~1;
    float ang = (float)k * __expf(-9.2103403719761836f * (float)iv / 512.0f);
    float pe = (c & 1) ? cosf(ang) : sinf(ang);
    float v = tab[i] + pe;
    emb0[i] = v;
    emb0h[i] = (_Float16)v;
}

// flat f32 -> f16
__global__ void cvt_h_kernel(const float* __restrict__ in, _Float16* __restrict__ out, int n) {
    int i = blockIdx.x * blockDim.x + threadIdx.x;
    if (i < n) out[i] = (_Float16)in[i];
}

// batched transpose + cvt: out[b][c*R + r] = (f16) in[b][r*C + c]
__global__ void transpose_h_kernel(const float* __restrict__ in, _Float16* __restrict__ outp,
                                   int R, int Cc, long long iSB, long long oSB, long long total) {
    long long i = (long long)blockIdx.x * blockDim.x + threadIdx.x;
    if (i >= total) return;
    long long rc = (long long)R * Cc;
    int b = (int)(i / rc);
    long long rem = i - (long long)b * rc;
    int c = (int)(rem / R);     // r fastest -> coalesced writes
    int r = (int)(rem - (long long)c * R);
    outp[(long long)b * oSB + (long long)c * R + r] = (_Float16)in[(long long)b * iSB + (long long)r * Cc + c];
}

// ---------------------------------------------------------------------------
// WMMA GEMM, all-f16 operands, B stored N x K (K contiguous).
//   C = alpha * (A @ B^T_layout) + bias ; writes f32 C and/or f16 Ch.
// Each wave: one 32x32 C tile (4 accumulators: 2 A frags x 2 B frags)
//   -> 8 b128 loads per 4 WMMAs in the steady-state loop.
// M mult of 32, N mult of 32, K mult of 32.
// Batch: grid.y = B*Hn, offsets bb*{a,b,c}SB + hh*{a,b,c}SH (element units).
// ---------------------------------------------------------------------------
__global__ void __launch_bounds__(256) gemm_h_kernel(
    const _Float16* __restrict__ A, int lda, long long aSB, long long aSH,
    const _Float16* __restrict__ Bt, int ldb, long long bSB, long long bSH,
    const float* __restrict__ bias,
    float* __restrict__ C, _Float16* __restrict__ Ch, int ldc, long long cSB, long long cSH,
    int M, int N, int K, int Hn, float alpha)
{
    int by = blockIdx.y;
    int bb = by / Hn;
    int hh = by - bb * Hn;
    const _Float16* Ab = A + (long long)bb * aSB + (long long)hh * aSH;
    const _Float16* Bb = Bt + (long long)bb * bSB + (long long)hh * bSH;

    int lane = threadIdx.x & 31;
    int hi   = lane >> 4;
    int l15  = lane & 15;

    int tilesN2 = N >> 5;
    int tiles   = (M >> 5) * tilesN2;
    int wavesTotal = gridDim.x * (blockDim.x >> 5);
    int wid = blockIdx.x * (blockDim.x >> 5) + (threadIdx.x >> 5);

    for (int t = wid; t < tiles; t += wavesTotal) {
        int tm = t / tilesN2;
        int tn = t - tm * tilesN2;
        int mBase = tm << 5;
        int nBase = tn << 5;

        v8f acc00 = {0.f,0.f,0.f,0.f,0.f,0.f,0.f,0.f};
        v8f acc01 = acc00, acc10 = acc00, acc11 = acc00;

        const _Float16* ap0 = Ab + (size_t)(mBase + l15) * lda + 8 * hi;
        const _Float16* ap1 = ap0 + (size_t)16 * lda;
        const _Float16* bp0 = Bb + (size_t)(nBase + l15) * ldb + 8 * hi;
        const _Float16* bp1 = bp0 + (size_t)16 * ldb;

        for (int k0 = 0; k0 < K; k0 += 32) {
            // A frags: VGPR0..3 = k0+8hi..+7 ; VGPR4..7 = k0+16+8hi..+7
            v16h a0 = combine16(*(const v8h*)(ap0 + k0), *(const v8h*)(ap0 + k0 + 16));
            v16h a1 = combine16(*(const v8h*)(ap1 + k0), *(const v8h*)(ap1 + k0 + 16));
            // B frags: VGPR0..3 = k0+8hi..+7 ; VGPR4..7 = k0+16+8hi..+7
            v16h b0 = combine16(*(const v8h*)(bp0 + k0), *(const v8h*)(bp0 + k0 + 16));
            v16h b1 = combine16(*(const v8h*)(bp1 + k0), *(const v8h*)(bp1 + k0 + 16));
            acc00 = __builtin_amdgcn_wmma_f32_16x16x32_f16(false, a0, false, b0, (short)0, acc00, false, false);
            acc01 = __builtin_amdgcn_wmma_f32_16x16x32_f16(false, a0, false, b1, (short)0, acc01, false, false);
            acc10 = __builtin_amdgcn_wmma_f32_16x16x32_f16(false, a1, false, b0, (short)0, acc10, false, false);
            acc11 = __builtin_amdgcn_wmma_f32_16x16x32_f16(false, a1, false, b1, (short)0, acc11, false, false);
        }

        int n0 = nBase + l15;
        float bv0 = bias ? bias[n0] : 0.0f;
        float bv1 = bias ? bias[n0 + 16] : 0.0f;
        long long cOff = (long long)bb * cSB + (long long)hh * cSH;
        #pragma unroll
        for (int v = 0; v < 8; ++v) {
            int m0 = mBase + 8 * hi + v;    // C layout: VGPR v -> M = v + 8*hi, N = l15
            int m1 = m0 + 16;
            float r00 = alpha * acc00[v] + bv0;
            float r01 = alpha * acc01[v] + bv1;
            float r10 = alpha * acc10[v] + bv0;
            float r11 = alpha * acc11[v] + bv1;
            size_t o0 = (size_t)cOff + (size_t)m0 * ldc + n0;
            size_t o1 = (size_t)cOff + (size_t)m1 * ldc + n0;
            if (C) {
                C[o0] = r00;  C[o0 + 16] = r01;
                C[o1] = r10;  C[o1 + 16] = r11;
            }
            if (Ch) {
                Ch[o0] = (_Float16)r00;  Ch[o0 + 16] = (_Float16)r01;
                Ch[o1] = (_Float16)r10;  Ch[o1 + 16] = (_Float16)r11;
            }
        }
    }
}

// ---------------------------------------------------------------------------
// Row softmax: read f32 scores, write f16 probabilities. One wave per row.
// ---------------------------------------------------------------------------
__global__ void __launch_bounds__(256) softmax_h_kernel(
    const float* __restrict__ S, _Float16* __restrict__ P, int R, int Ncol)
{
    int w = blockIdx.x * (blockDim.x >> 5) + (threadIdx.x >> 5);
    if (w >= R) return;
    int lane = threadIdx.x & 31;
    const float* row = S + (size_t)w * Ncol;
    _Float16* prow = P + (size_t)w * Ncol;
    float mx = -3.402823466e38f;
    for (int i = lane; i < Ncol; i += 32) mx = fmaxf(mx, row[i]);
    for (int o = 16; o > 0; o >>= 1) mx = fmaxf(mx, __shfl_xor(mx, o, 32));
    float sum = 0.0f;
    for (int i = lane; i < Ncol; i += 32) sum += __expf(row[i] - mx);
    for (int o = 16; o > 0; o >>= 1) sum += __shfl_xor(sum, o, 32);
    float inv = 1.0f / sum;
    for (int i = lane; i < Ncol; i += 32) prow[i] = (_Float16)(__expf(row[i] - mx) * inv);
}

// ---------------------------------------------------------------------------
// O[r,:] = LayerNorm(X[r,:] + Yres[(r % ymod),:]) ; writes f32 + f16. d=512.
// ---------------------------------------------------------------------------
__global__ void __launch_bounds__(256) add_ln_kernel(
    const float* __restrict__ X, const float* __restrict__ Yr, int ymod,
    const float* __restrict__ g, const float* __restrict__ be,
    float* __restrict__ O, _Float16* __restrict__ Oh, int R)
{
    int w = blockIdx.x * (blockDim.x >> 5) + (threadIdx.x >> 5);
    if (w >= R) return;
    int lane = threadIdx.x & 31;
    const float* x = X + (size_t)w * 512;
    const float* y = Yr + (size_t)(w % ymod) * 512;
    float vals[16];
    float s = 0.0f, s2 = 0.0f;
    #pragma unroll
    for (int i = 0; i < 16; ++i) {
        float v = x[lane + 32 * i] + y[lane + 32 * i];
        vals[i] = v; s += v; s2 += v * v;
    }
    for (int o = 16; o > 0; o >>= 1) { s += __shfl_xor(s, o, 32); s2 += __shfl_xor(s2, o, 32); }
    float mean = s * (1.0f / 512.0f);
    float var  = s2 * (1.0f / 512.0f) - mean * mean;
    float rstd = rsqrtf(var + 1e-5f);
    float* op = O + (size_t)w * 512;
    _Float16* oh = Oh + (size_t)w * 512;
    #pragma unroll
    for (int i = 0; i < 16; ++i) {
        int c = lane + 32 * i;
        float r = (vals[i] - mean) * rstd * g[c] + be[c];
        op[c] = r;
        oh[c] = (_Float16)r;
    }
}

// en[r] = sum_d E[r,d]^2, R rows of 512 (one wave per row)
__global__ void __launch_bounds__(256) row_sqnorm_kernel(
    const float* __restrict__ E, float* __restrict__ en, int R)
{
    int w = blockIdx.x * (blockDim.x >> 5) + (threadIdx.x >> 5);
    if (w >= R) return;
    int lane = threadIdx.x & 31;
    const float* r = E + (size_t)w * 512;
    float s = 0.0f;
    #pragma unroll
    for (int i = 0; i < 16; ++i) { float v = r[lane + 32 * i]; s += v * v; }
    for (int o = 16; o > 0; o >>= 1) s += __shfl_xor(s, o, 32);
    if (lane == 0) en[w] = s;
}

__global__ void zero_counts_kernel(float* __restrict__ counts) {
    if (threadIdx.x < 64) counts[threadIdx.x] = 0.0f;
}

// ---------------------------------------------------------------------------
// Per (b,t): argmin_k(en[b,k] - 2*dot[b,t,k]) -> idx, one-hot enc, z_q gather,
// per-row loss partial, code counts. Grid: 32768 x 64 threads.
// ---------------------------------------------------------------------------
__global__ void __launch_bounds__(64) quantize_kernel(
    const float* __restrict__ z, const float* __restrict__ emb,
    const float* __restrict__ dotv, const float* __restrict__ en,
    float* __restrict__ zq_out, float* __restrict__ enc_out,
    float* __restrict__ idx_out, float* __restrict__ lossP,
    float* __restrict__ counts)
{
    int bt = blockIdx.x;
    int b  = bt >> 11;
    int k  = threadIdx.x;
    __shared__ float sval[64];
    __shared__ int   sidx[64];
    __shared__ float sred[64];

    sval[k] = en[b * 64 + k] - 2.0f * dotv[(size_t)bt * 64 + k];
    sidx[k] = k;
    __syncthreads();
    for (int s = 32; s > 0; s >>= 1) {
        if (k < s) {
            float a = sval[k], bb = sval[k + s];
            int ia = sidx[k], ib = sidx[k + s];
            if (bb < a || (bb == a && ib < ia)) { sval[k] = bb; sidx[k] = ib; }
        }
        __syncthreads();
    }
    int kmin = sidx[0];

    enc_out[(size_t)bt * 64 + k] = (k == kmin) ? 1.0f : 0.0f;
    if (k == 0) idx_out[bt] = (float)kmin;

    const float* zr = z + (size_t)bt * 512;
    const float* er = emb + ((size_t)b * 64 + kmin) * 512;
    float ls = 0.0f;
    #pragma unroll
    for (int i = k; i < 512; i += 64) {
        float e = er[i];
        float d = e - zr[i];
        ls += d * d;
        zq_out[(size_t)bt * 512 + i] = e;   // straight-through: value == z_q
    }
    sred[k] = ls;
    __syncthreads();
    for (int s = 32; s > 0; s >>= 1) {
        if (k < s) sred[k] += sred[k + s];
        __syncthreads();
    }
    if (k == 0) {
        lossP[bt] = sred[0];
        atomicAdd(&counts[kmin], 1.0f);  // integer-valued: exact, deterministic
    }
}

__global__ void __launch_bounds__(256) finalize_kernel(
    const float* __restrict__ lossP, const float* __restrict__ counts,
    float* __restrict__ outLoss, float* __restrict__ outPerp)
{
    __shared__ float sh[256];
    int t = threadIdx.x;
    float s = 0.0f;
    for (int i = t; i < 32768; i += 256) s += lossP[i];
    sh[t] = s; __syncthreads();
    for (int st = 128; st > 0; st >>= 1) { if (t < st) sh[t] += sh[t + st]; __syncthreads(); }
    float lossSum = sh[0];
    __syncthreads();
    float ps = 0.0f;
    if (t < 64) {
        float em = counts[t] * (1.0f / 32768.0f);
        ps = em * logf(em + 1e-10f);
    }
    sh[t] = ps; __syncthreads();
    for (int st = 128; st > 0; st >>= 1) { if (t < st) sh[t] += sh[t + st]; __syncthreads(); }
    if (t == 0) {
        *outLoss = 1.5f * lossSum / (16.0f * 2048.0f * 512.0f);
        *outPerp = __expf(-sh[0]);
    }
}

__global__ void copyf_kernel(const float* __restrict__ s, float* __restrict__ d, int n) {
    int i = blockIdx.x * blockDim.x + threadIdx.x;
    if (i < n) d[i] = s[i];
}

// ---------------------------------------------------------------------------
extern "C" void kernel_launch(void* const* d_in, const int* in_sizes, int n_in,
                              void* d_out, int out_size, void* d_ws, size_t ws_size,
                              hipStream_t stream)
{
    (void)in_sizes; (void)n_in; (void)out_size; (void)ws_size;

    const float* x_enc = (const float*)d_in[0];
    const float* z     = (const float*)d_in[1];
    const float* tab   = (const float*)d_in[2];
    const float* sha_wq = (const float*)d_in[3];  const float* sha_bq = (const float*)d_in[4];
    const float* sha_wk = (const float*)d_in[5];  const float* sha_bk = (const float*)d_in[6];
    const float* sha_wv = (const float*)d_in[7];  const float* sha_bv = (const float*)d_in[8];
    const float* sha_wo = (const float*)d_in[9];  const float* sha_bo = (const float*)d_in[10];
    const float* e_wq = (const float*)d_in[11];   const float* e_bq = (const float*)d_in[12];
    const float* e_wk = (const float*)d_in[13];   const float* e_bk = (const float*)d_in[14];
    const float* e_wv = (const float*)d_in[15];   const float* e_bv = (const float*)d_in[16];
    const float* e_wo = (const float*)d_in[17];   const float* e_bo = (const float*)d_in[18];
    const float* eln_g = (const float*)d_in[19];  const float* eln_b = (const float*)d_in[20];
    const float* n_g = (const float*)d_in[21];    const float* n_b = (const float*)d_in[22];

    // -------- workspace carve (64B-aligned blocks) --------
    char* p = (char*)d_ws;
    auto allocF = [&](size_t n) { float* r = (float*)p; p += ((n * 4 + 63) / 64) * 64; return r; };
    auto allocH = [&](size_t n) { _Float16* r = (_Float16*)p; p += ((n * 2 + 63) / 64) * 64; return r; };

    float* emb0  = allocF(32768);
    float* Vb    = allocF(8388608);      // f32 V ; later aliased by zh (f16, 16.7M halves)
    float* Sc    = allocF(8388608);      // f32 scores ; later aliased by dotb
    float* ybuf  = allocF(524288);
    float* emb1  = allocF(524288);
    float* v2    = allocF(524288);
    float* y2    = allocF(524288);
    float* emb2  = allocF(524288);
    float* enb   = allocF(1024);
    float* counts= allocF(64);
    float* lossP = allocF(32768);

    _Float16* emb0h = allocH(32768);
    _Float16* xh    = allocH(8388608);
    _Float16* wqt   = allocH(262144);
    _Float16* wkt   = allocH(262144);
    _Float16* wvt   = allocH(262144);
    _Float16* wot   = allocH(262144);
    _Float16* ewqt  = allocH(262144);
    _Float16* ewkt  = allocH(262144);
    _Float16* ewvt  = allocH(262144);
    _Float16* ewot  = allocH(262144);
    _Float16* Qch   = allocH(32768);
    _Float16* Kbh   = allocH(8388608);
    _Float16* Vth   = allocH(8388608);   // V^T per b: (512 x 1024)
    _Float16* Sch   = allocH(8388608);   // probs (cross; reused for self)
    _Float16* ctxh  = allocH(524288);
    _Float16* emb1h = allocH(524288);
    _Float16* q2h   = allocH(524288);
    _Float16* k2h   = allocH(524288);
    _Float16* v2th  = allocH(524288);    // per b: (512 x 64)
    _Float16* ctx2h = allocH(524288);
    _Float16* emb2h = allocH(524288);

    _Float16* zh   = (_Float16*)Vb;      // alias: Vb dead after Vth transpose
    float*    dotb = Sc;                 // alias: Sc dead after self softmax

    // -------- output layout --------
    float* out      = (float*)d_out;
    float* out_zq   = out;               // 16*2048*512
    float* out_loss = out + 16777216;
    float* out_perp = out + 16777217;
    float* out_enc  = out + 16777218;    // 32768*64
    float* out_idx  = out + 18874370;    // 32768
    float* out_emb  = out + 18907138;    // 16*64*512

    // 0) operand preparation: emb0 (+f16), x_enc f16, transposed f16 weights
    build_emb0_kernel<<<128, 256, 0, stream>>>(tab, emb0, emb0h);
    cvt_h_kernel<<<32768, 256, 0, stream>>>(x_enc, xh, 8388608);
    transpose_h_kernel<<<1024, 256, 0, stream>>>(sha_wq, wqt, 512, 512, 0, 0, 262144);
    transpose_h_kernel<<<1024, 256, 0, stream>>>(sha_wk, wkt, 512, 512, 0, 0, 262144);
    transpose_h_kernel<<<1024, 256, 0, stream>>>(sha_wv, wvt, 512, 512, 0, 0, 262144);
    transpose_h_kernel<<<1024, 256, 0, stream>>>(sha_wo, wot, 512, 512, 0, 0, 262144);
    transpose_h_kernel<<<1024, 256, 0, stream>>>(e_wq, ewqt, 512, 512, 0, 0, 262144);
    transpose_h_kernel<<<1024, 256, 0, stream>>>(e_wk, ewkt, 512, 512, 0, 0, 262144);
    transpose_h_kernel<<<1024, 256, 0, stream>>>(e_wv, ewvt, 512, 512, 0, 0, 262144);
    transpose_h_kernel<<<1024, 256, 0, stream>>>(e_wo, ewot, 512, 512, 0, 0, 262144);

    // 1) projections: Q (batch-independent), K (f16 only), V (f32 for transpose)
    gemm_h_kernel<<<dim3(4, 1), 256, 0, stream>>>(
        emb0h, 512, 0, 0, wqt, 512, 0, 0, sha_bq,
        nullptr, Qch, 512, 0, 0, 64, 512, 512, 1, 1.0f);
    gemm_h_kernel<<<dim3(1024, 1), 256, 0, stream>>>(
        xh, 512, 0, 0, wkt, 512, 0, 0, sha_bk,
        nullptr, Kbh, 512, 0, 0, 16384, 512, 512, 1, 1.0f);
    gemm_h_kernel<<<dim3(1024, 1), 256, 0, stream>>>(
        xh, 512, 0, 0, wvt, 512, 0, 0, sha_bv,
        Vb, nullptr, 512, 0, 0, 16384, 512, 512, 1, 1.0f);
    // V^T per batch: (1024 x 512) -> (512 x 1024) f16
    transpose_h_kernel<<<32768, 256, 0, stream>>>(Vb, Vth, 1024, 512, 524288, 524288, 8388608);
    // z -> f16 (aliases Vb: must come after Vth)
    cvt_h_kernel<<<65536, 256, 0, stream>>>(z, zh, 16777216);

    // 2) cross attention: scores (f32) -> softmax (f16 probs) -> ctx (f16)
    gemm_h_kernel<<<dim3(8, 128), 256, 0, stream>>>(
        Qch, 512, 0, 64,
        Kbh, 512, 1024LL * 512, 64,
        nullptr,
        Sc, nullptr, 1024, 8LL * 65536, 65536,
        64, 1024, 64, 8, 0.125f);
    softmax_h_kernel<<<1024, 256, 0, stream>>>(Sc, Sch, 8192, 1024);
    gemm_h_kernel<<<dim3(1, 128), 256, 0, stream>>>(
        Sch, 1024, 8LL * 65536, 65536,
        Vth, 1024, 512LL * 1024, 64LL * 1024,
        nullptr,
        nullptr, ctxh, 512, 64LL * 512, 64,
        64, 64, 1024, 8, 1.0f);
    gemm_h_kernel<<<dim3(64, 1), 256, 0, stream>>>(
        ctxh, 512, 0, 0, wot, 512, 0, 0, sha_bo,
        ybuf, nullptr, 512, 0, 0, 1024, 512, 512, 1, 1.0f);
    add_ln_kernel<<<128, 256, 0, stream>>>(ybuf, emb0, 64, n_g, n_b, emb1, emb1h, 1024);

    // 3) self attention on emb1
    gemm_h_kernel<<<dim3(64, 1), 256, 0, stream>>>(
        emb1h, 512, 0, 0, ewqt, 512, 0, 0, e_bq,
        nullptr, q2h, 512, 0, 0, 1024, 512, 512, 1, 1.0f);
    gemm_h_kernel<<<dim3(64, 1), 256, 0, stream>>>(
        emb1h, 512, 0, 0, ewkt, 512, 0, 0, e_bk,
        nullptr, k2h, 512, 0, 0, 1024, 512, 512, 1, 1.0f);
    gemm_h_kernel<<<dim3(64, 1), 256, 0, stream>>>(
        emb1h, 512, 0, 0, ewvt, 512, 0, 0, e_bv,
        v2, nullptr, 512, 0, 0, 1024, 512, 512, 1, 1.0f);
    transpose_h_kernel<<<2048, 256, 0, stream>>>(v2, v2th, 64, 512, 32768, 32768, 524288);
    gemm_h_kernel<<<dim3(1, 128), 256, 0, stream>>>(
        q2h, 512, 64LL * 512, 64,
        k2h, 512, 64LL * 512, 64,
        nullptr,
        Sc, nullptr, 64, 8LL * 4096, 4096,
        64, 64, 64, 8, 0.125f);
    softmax_h_kernel<<<1024, 256, 0, stream>>>(Sc, Sch, 8192, 64);
    gemm_h_kernel<<<dim3(1, 128), 256, 0, stream>>>(
        Sch, 64, 8LL * 4096, 4096,
        v2th, 64, 512LL * 64, 64LL * 64,
        nullptr,
        nullptr, ctx2h, 512, 64LL * 512, 64,
        64, 64, 64, 8, 1.0f);
    gemm_h_kernel<<<dim3(64, 1), 256, 0, stream>>>(
        ctx2h, 512, 0, 0, ewot, 512, 0, 0, e_bo,
        y2, nullptr, 512, 0, 0, 1024, 512, 512, 1, 1.0f);
    add_ln_kernel<<<128, 256, 0, stream>>>(y2, emb1, 1024, eln_g, eln_b, emb2, emb2h, 1024);

    // 4) VQ: dot[b] (2048 x 64) = z_b @ emb2_b^T  (emb2h is naturally N x K)
    gemm_h_kernel<<<dim3(16, 16), 256, 0, stream>>>(
        zh, 512, 2048LL * 512, 0,
        emb2h, 512, 64LL * 512, 0,
        nullptr,
        dotb, nullptr, 64, 2048LL * 64, 0,
        2048, 64, 512, 1, 1.0f);
    row_sqnorm_kernel<<<128, 256, 0, stream>>>(emb2, enb, 1024);
    zero_counts_kernel<<<1, 64, 0, stream>>>(counts);
    quantize_kernel<<<32768, 64, 0, stream>>>(z, emb2, dotb, enb,
                                              out_zq, out_enc, out_idx, lossP, counts);
    finalize_kernel<<<1, 256, 0, stream>>>(lossP, counts, out_loss, out_perp);
    copyf_kernel<<<2048, 256, 0, stream>>>(emb2, out_emb, 524288);
}
